// RelationalTransformerEncoderLayer_73177652790000
// MI455X (gfx1250) — matrix operand order repaired and checked by reading, not compiled, not run on hardware
//
#include <hip/hip_runtime.h>

// ---------------------------------------------------------------------------
// RelationalTransformerEncoderLayer for gfx1250 (MI455X), wave32 + WMMA f16.
// B=4, N=1024, D=1024, H=16, DK=DV=64, FF=4096, NUM_EDGE_TYPES=16.
// GEMM tiles are staged with GLOBAL_LOAD_ASYNC_TO_LDS_B128 (ASYNCcnt path),
// double-buffered LDS, one barrier per k-tile, 8 WMMAs per wave per tile.
// ---------------------------------------------------------------------------

typedef __attribute__((ext_vector_type(16))) _Float16 v16h;
typedef __attribute__((ext_vector_type(8)))  _Float16 v8h;
typedef __attribute__((ext_vector_type(8)))  float    v8f;

#define LDS_STRIDE 40   // 32 + 8 pad halves; 80 bytes -> multiple of 16B

// CDNA5 async L2->LDS copy (16B per lane), tracked by ASYNCcnt.
__device__ __forceinline__ void async_ld_b128(unsigned lds_addr, const void* gaddr) {
    asm volatile("global_load_async_to_lds_b128 %0, %1, off"
                 :: "v"(lds_addr), "v"(gaddr) : "memory");
}
__device__ __forceinline__ void wait_asynccnt0() {
    asm volatile("s_wait_asynccnt 0x0" ::: "memory");
}

// ---------------------------------------------------------------------------
// Batched GEMM: C[z] = A[z] (MxK, row-major) * B[z]^T  (B given NxK row-major)
// f16 inputs, f32 accumulate via v_wmma_f32_16x16x32_f16.
// storeMode: 0 = plain        off = z*sCz + row*ldc + col
//            1 = split-head   rows=(b,n) cols=(h,d) -> (B,H,N,64)
//            2 = merge-head   z=(b,h), rows=q, cols=d -> (B,N,H*64)
//            3 = split-head-T rows=(b,n) cols=(h,d) -> (B,H,64,N)
// ---------------------------------------------------------------------------
__global__ __launch_bounds__(256) void wmma_gemm_kernel(
    const _Float16* __restrict__ A, long sAz, int lda,
    const _Float16* __restrict__ Bm, long sBz, int ldb,
    float* __restrict__ Cf, _Float16* __restrict__ Ch, long sCz, int ldc,
    const float* __restrict__ bias, int relu, int storeMode,
    int Nd, int K)
{
    __shared__ __attribute__((aligned(16))) _Float16 As[2][128 * LDS_STRIDE];
    __shared__ __attribute__((aligned(16))) _Float16 Bs[2][128 * LDS_STRIDE];

    const int tid  = threadIdx.x;
    const int lane = tid & 31;
    const int wid  = tid >> 5;
    const int wm   = wid >> 2;          // 0..1  (64-row band)
    const int wn   = wid & 3;           // 0..3  (32-col band)
    const int z    = blockIdx.z;
    const int m0   = blockIdx.y * 128;
    const int n0   = blockIdx.x * 128;

    const _Float16* Ab = A  + (long)z * sAz;
    const _Float16* Bb = Bm + (long)z * sBz;

    const int lh = lane >> 4;           // lane-half
    const int ll = lane & 15;

    // fixed per-thread staging coordinates: 512 16B-vectors per 128x32 tile
    const int r0 = tid >> 2;            // 0..63
    const int c0 = (tid & 3) << 3;      // 0,8,16,24
    const _Float16* Aptr0 = Ab + (long)(m0 + r0)      * lda + c0;
    const _Float16* Aptr1 = Ab + (long)(m0 + r0 + 64) * lda + c0;
    const bool bok0 = (n0 + r0)      < Nd;
    const bool bok1 = (n0 + r0 + 64) < Nd;
    const _Float16* Bptr0 = Bb + (long)(n0 + r0)      * ldb + c0;
    const _Float16* Bptr1 = Bb + (long)(n0 + r0 + 64) * ldb + c0;

    // per-lane LDS destination byte addresses (low 32 bits of flat address)
    unsigned lA0[2], lA1[2], lB0[2], lB1[2];
#pragma unroll
    for (int buf = 0; buf < 2; ++buf) {
        lA0[buf] = (unsigned)(uintptr_t)&As[buf][r0 * LDS_STRIDE + c0];
        lA1[buf] = (unsigned)(uintptr_t)&As[buf][(r0 + 64) * LDS_STRIDE + c0];
        lB0[buf] = (unsigned)(uintptr_t)&Bs[buf][r0 * LDS_STRIDE + c0];
        lB1[buf] = (unsigned)(uintptr_t)&Bs[buf][(r0 + 64) * LDS_STRIDE + c0];
    }

    // one-time zero-fill of OOB B rows (async loads never touch them)
    v8h z8 = {};
    if (!bok0) {
        *(v8h*)&Bs[0][r0 * LDS_STRIDE + c0] = z8;
        *(v8h*)&Bs[1][r0 * LDS_STRIDE + c0] = z8;
    }
    if (!bok1) {
        *(v8h*)&Bs[0][(r0 + 64) * LDS_STRIDE + c0] = z8;
        *(v8h*)&Bs[1][(r0 + 64) * LDS_STRIDE + c0] = z8;
    }

    auto issueTile = [&](int k0, int buf) {
        async_ld_b128(lA0[buf], Aptr0 + k0);
        async_ld_b128(lA1[buf], Aptr1 + k0);
        if (bok0) async_ld_b128(lB0[buf], Bptr0 + k0);
        if (bok1) async_ld_b128(lB1[buf], Bptr1 + k0);
    };

    v8f zero = {};
    v8f acc[4][2];
#pragma unroll
    for (int i = 0; i < 4; ++i)
#pragma unroll
        for (int j = 0; j < 2; ++j) acc[i][j] = zero;

    const int nk = K >> 5;
    issueTile(0, 0);
    wait_asynccnt0();
    __syncthreads();

    for (int kt = 0; kt < nk; ++kt) {
        const int cur = kt & 1;
        if (kt + 1 < nk) {
            issueTile((kt + 1) << 5, cur ^ 1);       // L2 -> LDS, no VGPR trip
            if (kt + 2 < nk) {
                int kp = (kt + 2) << 5;
                __builtin_prefetch(Aptr0 + kp, 0, 1);  // global_prefetch_b8
                __builtin_prefetch(Bptr0 + kp, 0, 1);
            }
        }

        // ---- per-lane fragment gathers (ISA 7.12.2 layouts) ----
        union Frag { v16h v; v8h h[2]; };
        Frag afr[4], bfr[2];
#pragma unroll
        for (int tm = 0; tm < 4; ++tm) {
            int row  = wm * 64 + tm * 16 + ll;
            int base = row * LDS_STRIDE + (lh << 3);         // +8 halves upper lanes
            afr[tm].h[0] = *(const v8h*)&As[cur][base];      // K = 0..7  (+8)
            afr[tm].h[1] = *(const v8h*)&As[cur][base + 16]; // K = 16..23(+8)
        }
#pragma unroll
        for (int tn = 0; tn < 2; ++tn) {
            int col  = wn * 32 + tn * 16 + ll;
            int base = col * LDS_STRIDE + (lh << 3);         // +8 halves upper lanes
            bfr[tn].h[0] = *(const v8h*)&Bs[cur][base];      // K = 0..7  (+8)
            bfr[tn].h[1] = *(const v8h*)&Bs[cur][base + 16]; // K = 16..23(+8)
        }
#pragma unroll
        for (int tm = 0; tm < 4; ++tm)
#pragma unroll
            for (int tn = 0; tn < 2; ++tn)
                acc[tm][tn] = __builtin_amdgcn_wmma_f32_16x16x32_f16(
                    false, afr[tm].v, false, bfr[tn].v,
                    (short)0, acc[tm][tn], false, false);

        if (kt + 1 < nk) {
            wait_asynccnt0();            // own-wave asyncs landed in LDS
            __syncthreads();             // everyone's landed
        }
    }

    // ---- epilogue ----
#pragma unroll
    for (int tm = 0; tm < 4; ++tm) {
#pragma unroll
        for (int tn = 0; tn < 2; ++tn) {
            int colL = n0 + wn * 32 + tn * 16 + ll;
            if (colL >= Nd) continue;
            float bv = bias ? bias[colL] : 0.0f;
#pragma unroll
            for (int r = 0; r < 8; ++r) {
                int rowL = m0 + wm * 64 + tm * 16 + (lh << 3) + r;
                float val = acc[tm][tn][r] + bv;
                if (relu) val = val > 0.0f ? val : 0.0f;
                long off;
                if (storeMode == 0) {
                    off = (long)z * sCz + (long)rowL * ldc + colL;
                } else if (storeMode == 1) {
                    int b = rowL >> 10, n = rowL & 1023;
                    int h = colL >> 6,  d = colL & 63;
                    off = ((long)((b * 16 + h) * 1024 + n)) * 64 + d;
                } else if (storeMode == 2) {
                    int b = z >> 4, h = z & 15;
                    off = ((long)((b * 1024 + rowL) * 16 + h)) * 64 + colL;
                } else {
                    int b = rowL >> 10, n = rowL & 1023;
                    int h = colL >> 6,  d = colL & 63;
                    off = ((long)((b * 16 + h) * 64 + d)) * 1024 + n;
                }
                if (Cf) Cf[off] = val;
                if (Ch) Ch[off] = (_Float16)val;
            }
        }
    }
}

// ---------------------------------------------------------------------------
// float -> f16 conversion (n multiple of 4)
// ---------------------------------------------------------------------------
__global__ __launch_bounds__(256) void cvt_kernel(const float* __restrict__ in,
                                                  _Float16* __restrict__ out, long n)
{
    long i = ((long)blockIdx.x * 256 + threadIdx.x) * 4;
    if (i >= n) return;
    float4 v = *(const float4*)(in + i);
    out[i + 0] = (_Float16)v.x;
    out[i + 1] = (_Float16)v.y;
    out[i + 2] = (_Float16)v.z;
    out[i + 3] = (_Float16)v.w;
}

// ---------------------------------------------------------------------------
// Transpose + convert: in (R x C fp32 row-major) -> out (C x R f16 row-major).
// 32x32 LDS tiles, coalesced both directions. R, C multiples of 32.
// ---------------------------------------------------------------------------
__global__ __launch_bounds__(256) void cvtT_kernel(const float* __restrict__ in,
                                                   _Float16* __restrict__ out,
                                                   int R, int C)
{
    __shared__ _Float16 tile[32][33];
    int bx = blockIdx.x * 32;           // col base in input
    int by = blockIdx.y * 32;           // row base in input
    int tx = threadIdx.x;               // 0..31
    int ty = threadIdx.y;               // 0..7
#pragma unroll
    for (int i = 0; i < 32; i += 8)
        tile[ty + i][tx] = (_Float16)in[(long)(by + ty + i) * C + bx + tx];
    __syncthreads();
#pragma unroll
    for (int i = 0; i < 32; i += 8)
        out[(long)(bx + ty + i) * R + by + tx] = tile[tx][ty + i];
}

// ---------------------------------------------------------------------------
// Edge-bias scatter: one thread per (edge, head), 64-wide dot, atomicAdd into
// scores[(b,h,q=tgt,k=src)].
// ---------------------------------------------------------------------------
__global__ __launch_bounds__(256) void edge_bias_kernel(
    const int* __restrict__ edges, const int* __restrict__ etypes,
    const float* __restrict__ ebemb, const _Float16* __restrict__ q,
    float* __restrict__ scores, int E_)
{
    int t = blockIdx.x * 256 + threadIdx.x;
    if (t >= E_ * 16) return;
    int e  = t >> 4, h = t & 15;
    int b  = edges[e * 3 + 0];
    int sN = edges[e * 3 + 1];   // source node -> k index
    int tg = edges[e * 3 + 2];   // target node -> q index
    const float*    eb = ebemb + (long)etypes[e] * 1024 + h * 64;
    const _Float16* qp = q + ((long)((b * 16 + h) * 1024 + tg)) * 64;
    float acc = 0.0f;
#pragma unroll 8
    for (int d = 0; d < 64; ++d) acc += eb[d] * (float)qp[d];
    atomicAdd(&scores[((long)((b * 16 + h) * 1024 + tg)) * 1024 + sN], acc);
}

// ---------------------------------------------------------------------------
// Row softmax over k (1024), scale 1/8, mask, write f16 probs in-place into
// the first half of each fp32 row.  gridDim.x = B*H*N = 65536.
// ---------------------------------------------------------------------------
__global__ __launch_bounds__(256) void softmax_kernel(float* __restrict__ scores,
                                                      const unsigned char* __restrict__ mask)
{
    __shared__ float sbuf[256];
    long rowbase = (long)blockIdx.x << 10;
    float* row = scores + rowbase;
    int b = blockIdx.x >> 14;                  // row = ((b*16+h)*1024 + q)
    const unsigned char* mrow = mask + (long)b * 1024;
    int t = threadIdx.x;

    float x[4];
    float m = -3.0e38f;
#pragma unroll
    for (int i = 0; i < 4; ++i) {
        int k = t + i * 256;
        float v = row[k] * 0.125f;             // 1/sqrt(64)
        if (mrow[k]) v = -3.0e38f;
        x[i] = v;
        m = fmaxf(m, v);
    }
    sbuf[t] = m; __syncthreads();
    for (int s = 128; s > 0; s >>= 1) { if (t < s) sbuf[t] = fmaxf(sbuf[t], sbuf[t + s]); __syncthreads(); }
    m = sbuf[0]; __syncthreads();

    float sum = 0.0f;
#pragma unroll
    for (int i = 0; i < 4; ++i) { x[i] = __expf(x[i] - m); sum += x[i]; }
    sbuf[t] = sum; __syncthreads();
    for (int s = 128; s > 0; s >>= 1) { if (t < s) sbuf[t] += sbuf[t + s]; __syncthreads(); }
    float inv = 1.0f / sbuf[0];

    _Float16* prow = (_Float16*)row;           // aliases first 2KB of the 4KB row
#pragma unroll
    for (int i = 0; i < 4; ++i) { int k = t + i * 256; prow[k] = (_Float16)(x[i] * inv); }
}

// ---------------------------------------------------------------------------
// y = LayerNorm(a + c) * g + be   (row length 1024), fp32 out + optional f16.
// ---------------------------------------------------------------------------
__global__ __launch_bounds__(256) void add_ln_kernel(
    const float* __restrict__ a, const float* __restrict__ c,
    const float* __restrict__ g, const float* __restrict__ be,
    float* __restrict__ outf, _Float16* __restrict__ outh)
{
    __shared__ float sbuf[256];
    long base = (long)blockIdx.x << 10;
    int t = threadIdx.x;

    float x[4];
    float s = 0.0f;
#pragma unroll
    for (int i = 0; i < 4; ++i) { int k = t + i * 256; x[i] = a[base + k] + c[base + k]; s += x[i]; }
    sbuf[t] = s; __syncthreads();
    for (int st = 128; st > 0; st >>= 1) { if (t < st) sbuf[t] += sbuf[t + st]; __syncthreads(); }
    float mean = sbuf[0] * (1.0f / 1024.0f);
    __syncthreads();

    float vs = 0.0f;
#pragma unroll
    for (int i = 0; i < 4; ++i) { float d = x[i] - mean; vs += d * d; }
    sbuf[t] = vs; __syncthreads();
    for (int st = 128; st > 0; st >>= 1) { if (t < st) sbuf[t] += sbuf[t + st]; __syncthreads(); }
    float rstd = rsqrtf(sbuf[0] * (1.0f / 1024.0f) + 1e-5f);

#pragma unroll
    for (int i = 0; i < 4; ++i) {
        int k = t + i * 256;
        float y = (x[i] - mean) * rstd * g[k] + be[k];
        if (outf) outf[base + k] = y;
        if (outh) outh[base + k] = (_Float16)y;
    }
}

// ---------------------------------------------------------------------------
extern "C" void kernel_launch(void* const* d_in, const int* in_sizes, int n_in,
                              void* d_out, int out_size, void* d_ws, size_t ws_size,
                              hipStream_t stream)
{
    const float* src   = (const float*)d_in[0];
    const float* Wq    = (const float*)d_in[1];
    const float* Wk    = (const float*)d_in[2];
    const float* Wv    = (const float*)d_in[3];
    const float* Wo    = (const float*)d_in[4];
    const float* bo    = (const float*)d_in[5];
    const float* ebemb = (const float*)d_in[6];
    const float* W1    = (const float*)d_in[7];
    const float* b1    = (const float*)d_in[8];
    const float* W2    = (const float*)d_in[9];
    const float* b2    = (const float*)d_in[10];
    const float* ln1g  = (const float*)d_in[11];
    const float* ln1b  = (const float*)d_in[12];
    const float* ln2g  = (const float*)d_in[13];
    const float* ln2b  = (const float*)d_in[14];
    const unsigned char* mask = (const unsigned char*)d_in[15];
    const int* edges   = (const int*)d_in[16];
    const int* etypes  = (const int*)d_in[17];
    const int  E_      = in_sizes[17];
    float* out         = (float*)d_out;

    // ---- workspace carve-up ----
    char* ws = (char*)d_ws;
    size_t off = 0;
    auto alloc = [&](size_t bytes) -> void* {
        void* p = ws + off;
        off += (bytes + 255) & ~(size_t)255;
        return p;
    };
    _Float16* srcb  = (_Float16*)alloc(4194304ull * 2);   // src f16 (B*N, D)
    _Float16* wqbT  = (_Float16*)alloc(1048576ull * 2);   // (H*DK, D)
    _Float16* wkbT  = (_Float16*)alloc(1048576ull * 2);
    _Float16* wvbT  = (_Float16*)alloc(1048576ull * 2);
    _Float16* wobT  = (_Float16*)alloc(1048576ull * 2);   // (D, H*DV)
    _Float16* w1bT  = (_Float16*)alloc(4194304ull * 2);   // (FF, D)
    _Float16* w2bT  = (_Float16*)alloc(4194304ull * 2);   // (D, FF)
    _Float16* qb    = (_Float16*)alloc(4194304ull * 2);   // (B,H,N,64)
    _Float16* kb    = (_Float16*)alloc(4194304ull * 2);   // (B,H,N,64)
    _Float16* vTb   = (_Float16*)alloc(4194304ull * 2);   // (B,H,64,N)
    float*    sc    = (float*)   alloc(67108864ull * 4);  // (B*H,N,N) scores / f16 probs aliased
    _Float16* ctxb  = (_Float16*)alloc(4194304ull * 2);   // (B*N, H*64)
    float*    attn  = (float*)   alloc(4194304ull * 4);
    float*    xf    = (float*)   alloc(4194304ull * 4);
    _Float16* xb    = (_Float16*)alloc(4194304ull * 2);
    _Float16* ffh   = (_Float16*)alloc(16777216ull * 2);  // (B*N, FF)
    float*    fft   = (float*)   alloc(4194304ull * 4);

    auto gemm = [&](const _Float16* A, long sAz, int lda,
                    const _Float16* Bmat, long sBz, int ldb,
                    float* Cf, _Float16* Ch, long sCz, int ldc,
                    const float* bias, int relu, int mode,
                    int M, int Nd, int K, int batch) {
        dim3 grid((Nd + 127) / 128, M / 128, batch);
        wmma_gemm_kernel<<<grid, 256, 0, stream>>>(A, sAz, lda, Bmat, sBz, ldb,
                                                   Cf, Ch, sCz, ldc, bias, relu, mode, Nd, K);
    };

    // ---- conversions (weights transposed to N x K) ----
    cvt_kernel <<<4096, 256, 0, stream>>>(src, srcb, 4194304);
    dim3 tb(32, 8);
    cvtT_kernel<<<dim3(32, 32),  tb, 0, stream>>>(Wq, wqbT, 1024, 1024);
    cvtT_kernel<<<dim3(32, 32),  tb, 0, stream>>>(Wk, wkbT, 1024, 1024);
    cvtT_kernel<<<dim3(32, 32),  tb, 0, stream>>>(Wv, wvbT, 1024, 1024);
    cvtT_kernel<<<dim3(32, 32),  tb, 0, stream>>>(Wo, wobT, 1024, 1024);
    cvtT_kernel<<<dim3(128, 32), tb, 0, stream>>>(W1, w1bT, 1024, 4096);
    cvtT_kernel<<<dim3(32, 128), tb, 0, stream>>>(W2, w2bT, 4096, 1024);

    // ---- QKV projections: q,k split-head (B,H,N,64); v split-head-T (B,H,64,N) ----
    gemm(srcb, 0, 1024, wqbT, 0, 1024, nullptr, qb,  0, 0, nullptr, 0, 1, 4096, 1024, 1024, 1);
    gemm(srcb, 0, 1024, wkbT, 0, 1024, nullptr, kb,  0, 0, nullptr, 0, 1, 4096, 1024, 1024, 1);
    gemm(srcb, 0, 1024, wvbT, 0, 1024, nullptr, vTb, 0, 0, nullptr, 0, 3, 4096, 1024, 1024, 1);

    // ---- scores[z=(b,h)] = Q_bh (1024x64) @ K_bh^T ----
    gemm(qb, 65536, 64, kb, 65536, 64, sc, nullptr, 1048576, 1024,
         nullptr, 0, 0, 1024, 1024, 64, 64);

    // ---- sparse edge bias scatter ----
    edge_bias_kernel<<<(E_ * 16 + 255) / 256, 256, 0, stream>>>(edges, etypes, ebemb, qb, sc, E_);

    // ---- scale + mask + softmax; probs stored f16 in-place ----
    softmax_kernel<<<65536, 256, 0, stream>>>(sc, mask);

    // ---- ctx[z=(b,h)] = probs (1024x1024 f16, lda=2048) @ vT_bh^T (64x1024) ----
    gemm((const _Float16*)sc, 2097152, 2048, vTb, 65536, 1024,
         nullptr, ctxb, 0, 0, nullptr, 0, 2, 1024, 64, 1024, 64);

    // ---- attn_out = ctx @ Wo + bo ----
    gemm(ctxb, 0, 1024, wobT, 0, 1024, attn, nullptr, 0, 1024, bo, 0, 0, 4096, 1024, 1024, 1);

    // ---- x = LN(src + attn_out) ----
    add_ln_kernel<<<4096, 256, 0, stream>>>(src, attn, ln1g, ln1b, xf, xb);

    // ---- ff hidden = relu(x @ W1 + b1) ----
    gemm(xb, 0, 1024, w1bT, 0, 1024, nullptr, ffh, 0, 4096, b1, 1, 0, 4096, 4096, 1024, 1);

    // ---- ff out = h @ W2 + b2 ----
    gemm(ffh, 0, 4096, w2bT, 0, 4096, fft, nullptr, 0, 1024, b2, 0, 0, 4096, 1024, 4096, 1);

    // ---- out = LN(x + ff) ----
    add_ln_kernel<<<4096, 256, 0, stream>>>(xf, fft, ln2g, ln2b, out, nullptr);
}
